// GNN_46411416600696
// MI455X (gfx1250) — matrix operand order repaired and checked by reading, not compiled, not run on hardware
//
#include <hip/hip_runtime.h>
#include <hip/hip_bf16.h>
#include <stddef.h>

// ---------- types ----------
typedef __attribute__((ext_vector_type(16))) _Float16 v16h;
typedef __attribute__((ext_vector_type(8)))  _Float16 v8h;
typedef __attribute__((ext_vector_type(4)))  _Float16 v4h;
typedef __attribute__((ext_vector_type(8)))  float    v8f;
typedef __attribute__((ext_vector_type(4)))  int      v4i;

#define D_IN    128
#define HIDDEN  512
#define NCLASS  2
#define KCHUNK  64

// gfx1250 async global->LDS path (guarded; falls back to load+ds_store staging)
#if defined(__AMDGCN__) && __has_builtin(__builtin_amdgcn_global_load_async_to_lds_b128) && __has_builtin(__builtin_amdgcn_s_wait_asynccnt)
#define USE_ASYNC_LDS 1
typedef __attribute__((address_space(1))) v4i as1_v4i;   // global int4
typedef __attribute__((address_space(3))) v4i as3_v4i;   // LDS int4
#else
#define USE_ASYNC_LDS 0
#endif

// ---------- small utility kernels ----------

__global__ void k_deg_init(float* __restrict__ deg, int n) {
    int i = blockIdx.x * blockDim.x + threadIdx.x;
    if (i < n) deg[i] = 1.0f;   // self-loop contributes 1 to every node's degree
}

__global__ void k_deg_edges(const int* __restrict__ row, float* __restrict__ deg, int E) {
    int e = blockIdx.x * blockDim.x + threadIdx.x;
    if (e < E) atomicAdd(&deg[row[e]], 1.0f);
}

__global__ void k_dinv(float* __restrict__ deg, int n) {
    int i = blockIdx.x * blockDim.x + threadIdx.x;
    if (i < n) {
        float d = deg[i];                 // always >= 1 (self loop)
        deg[i] = rsqrtf(d);               // in-place: buffer becomes dinv
    }
}

__global__ void k_cvt_f16(const float* __restrict__ src, _Float16* __restrict__ dst, int n) {
    int i = blockIdx.x * blockDim.x + threadIdx.x;
    if (i < n) dst[i] = (_Float16)src[i];
}

// W [K,N] fp32 -> Wt [N,K] f16 (so B fragments are contiguous in K)
__global__ void k_cvt_w_t(const float* __restrict__ W, _Float16* __restrict__ Wt, int K, int N) {
    int i = blockIdx.x * blockDim.x + threadIdx.x;
    if (i < K * N) {
        int k = i / N, n = i % N;
        Wt[(size_t)n * K + k] = (_Float16)W[(size_t)k * N + n];
    }
}

// ---------- WMMA GEMM:  C[M,N](f16) = A[M,K](f16) @ Bt[N,K](f16), fp32 accum ----------
// Block = 8 waves sharing one 64-column N-group across 8 consecutive M-tiles.
// The B strip for each 64-wide K chunk is staged into LDS with
// global_load_async_to_lds_b128, double-buffered so chunk i+1's async loads fly
// while chunk i feeds the WMMAs. Each wave owns a 16x64 strip (4 accumulators).
__global__ __launch_bounds__(256) void k_gemm_f16(
    const _Float16* __restrict__ A,
    const _Float16* __restrict__ Bt,
    _Float16* __restrict__ C,
    int M, int N, int K)
{
    __shared__ _Float16 Bs[2][64 * KCHUNK];   // 2 x 8 KB ping-pong

    const int lane    = threadIdx.x & 31;
    const int w       = threadIdx.x >> 5;             // wave in block (0..7)
    const int ngroups = N >> 6;                       // N / 64
    const int mtiles  = (M + 15) >> 4;
    const int mg      = blockIdx.x / ngroups;
    const int ng      = blockIdx.x % ngroups;
    const int mt      = mg * 8 + w;                   // may exceed mtiles (ragged tail)

    const int hi  = lane >> 4;
    const int l15 = lane & 15;
    int m = mt * 16 + l15;
    if (m >= M) m = M - 1;                            // clamp; padded rows never stored

    v8f acc0 = {}, acc1 = {}, acc2 = {}, acc3 = {};

    const _Float16* arow = A + (size_t)m * K;
    const _Float16* bsrc = Bt + (size_t)(ng * 64) * K;   // 64 weight rows of length K
    const int t = threadIdx.x;

    // cooperative stage of one 64-col x 64-K strip: 512 x 16B pieces, 2 per thread
    auto stage = [&](int k0, int buf) {
        #pragma unroll
        for (int s = 0; s < 2; ++s) {
            int ch = t + s * 256;         // 0..511
            int c  = ch >> 3;             // column within group
            int p  = ch & 7;              // 16B piece within row
            const _Float16* g = bsrc + (size_t)c * K + k0 + p * 8;
            _Float16*       l = &Bs[buf][c * KCHUNK + p * 8];
#if USE_ASYNC_LDS
            __builtin_amdgcn_global_load_async_to_lds_b128(
                (as1_v4i*)g, (as3_v4i*)l, 0, 0);
#else
            *(v8h*)l = *(const v8h*)g;
#endif
        }
    };

    const int nchunks = K / KCHUNK;
    stage(0, 0);                          // prologue: chunk 0 in flight

    for (int ci = 0; ci < nchunks; ++ci) {
        const int k0 = ci * KCHUNK;
#if USE_ASYNC_LDS
        __builtin_amdgcn_s_wait_asynccnt(0);   // my share of chunk ci has landed
#endif
        __syncthreads();                       // all waves' shares visible; prev reads done

        if (ci + 1 < nchunks)                  // overlap: next chunk loads vs this compute
            stage(k0 + KCHUNK, (ci + 1) & 1);

        const _Float16* Bbuf = Bs[ci & 1];

        #pragma unroll
        for (int kk = 0; kk < KCHUNK; kk += 32) {
            // A fragment: lane holds row m; K chunk [ks+hi*16,+16)
            const int ks = k0 + kk;
            v16h a = *(const v16h*)(arow + ks + hi * 16);

            // B fragments from LDS: lane holds column; 16 contiguous K per half-wave
            const _Float16* bbase = &Bbuf[l15 * KCHUNK + kk + hi * 16];
            v16h b0 = *(const v16h*)(bbase);
            v16h b1 = *(const v16h*)(bbase + 16 * KCHUNK);
            v16h b2 = *(const v16h*)(bbase + 32 * KCHUNK);
            v16h b3 = *(const v16h*)(bbase + 48 * KCHUNK);

            acc0 = __builtin_amdgcn_wmma_f32_16x16x32_f16(false, a, false, b0, (short)0, acc0, false, false);
            acc1 = __builtin_amdgcn_wmma_f32_16x16x32_f16(false, a, false, b1, (short)0, acc1, false, false);
            acc2 = __builtin_amdgcn_wmma_f32_16x16x32_f16(false, a, false, b2, (short)0, acc2, false, false);
            acc3 = __builtin_amdgcn_wmma_f32_16x16x32_f16(false, a, false, b3, (short)0, acc3, false, false);
        }
    }

    // C/D layout: VGPR v -> row (v + 8*hi), col = lane&15
    if (mt < mtiles) {
        const int colBase = ng * 64 + l15;
        #pragma unroll
        for (int v = 0; v < 8; ++v) {
            int row = mt * 16 + hi * 8 + v;
            if (row < M) {
                _Float16* out = C + (size_t)row * N + colBase;
                out[0]  = (_Float16)acc0[v];
                out[16] = (_Float16)acc1[v];
                out[32] = (_Float16)acc2[v];
                out[48] = (_Float16)acc3[v];
            }
        }
    }
}

// ---------- edge aggregation: agg[row] += lin[col] * dinv[row]*dinv[col] ----------
// one block per edge; 128 threads x 4 features = 512
__global__ __launch_bounds__(128) void k_agg_edges(
    const _Float16* __restrict__ lin,
    const float* __restrict__ dinv,
    const int* __restrict__ row,
    const int* __restrict__ col,
    float* __restrict__ agg,
    int E)
{
    int e = blockIdx.x;
    if (e >= E) return;
    int r  = row[e];
    int cc = col[e];
    float wgt = dinv[r] * dinv[cc];
    int f = threadIdx.x * 4;
    v4h v = *(const v4h*)(lin + (size_t)cc * HIDDEN + f);
    float* dst = agg + (size_t)r * HIDDEN + f;
    atomicAdd(dst + 0, (float)v[0] * wgt);
    atomicAdd(dst + 1, (float)v[1] * wgt);
    atomicAdd(dst + 2, (float)v[2] * wgt);
    atomicAdd(dst + 3, (float)v[3] * wgt);
}

// ---------- finalize: h = relu(agg + selfloop + bias) -> f16 ----------
__global__ __launch_bounds__(256) void k_finalize(
    const float* __restrict__ agg,
    const _Float16* __restrict__ lin,
    const float* __restrict__ dinv,
    const float* __restrict__ bias,
    _Float16* __restrict__ hout,
    int NN)
{
    int i = blockIdx.x;
    if (i >= NN) return;
    float d2 = dinv[i] * dinv[i];           // self-loop norm = dinv[i]^2
    #pragma unroll
    for (int s = 0; s < 2; ++s) {
        int f = threadIdx.x + s * 256;
        float v = agg[(size_t)i * HIDDEN + f]
                + d2 * (float)lin[(size_t)i * HIDDEN + f]
                + bias[f];
        hout[(size_t)i * HIDDEN + f] = (_Float16)(v > 0.0f ? v : 0.0f);
    }
}

// ---------- sum pooling per graph ----------
__global__ __launch_bounds__(256) void k_pool(
    const _Float16* __restrict__ h,
    const int* __restrict__ ngi,
    float* __restrict__ pool,
    int NN)
{
    int i = blockIdx.x;
    if (i >= NN) return;
    int g = ngi[i];
    #pragma unroll
    for (int s = 0; s < 2; ++s) {
        int f = threadIdx.x + s * 256;
        atomicAdd(&pool[(size_t)g * HIDDEN + f], (float)h[(size_t)i * HIDDEN + f]);
    }
}

// ---------- final FC (tiny, fp32) ----------
__global__ void k_fc(const float* __restrict__ pool,
                     const float* __restrict__ Wfc,
                     const float* __restrict__ bfc,
                     float* __restrict__ out, int G)
{
    int idx = blockIdx.x * blockDim.x + threadIdx.x;
    if (idx >= G * NCLASS) return;
    int g = idx / NCLASS, c = idx % NCLASS;
    float s = bfc[c];
    for (int k = 0; k < HIDDEN; ++k)
        s += pool[(size_t)g * HIDDEN + k] * Wfc[(size_t)k * NCLASS + c];
    out[idx] = s;
}

// ---------- host launch ----------
extern "C" void kernel_launch(void* const* d_in, const int* in_sizes, int n_in,
                              void* d_out, int out_size, void* d_ws, size_t ws_size,
                              hipStream_t stream)
{
    const float* x    = (const float*)d_in[0];
    const int*   eidx = (const int*)  d_in[1];
    const int*   ngi  = (const int*)  d_in[2];
    const float* W0   = (const float*)d_in[3];
    const float* b0   = (const float*)d_in[4];
    const float* W1   = (const float*)d_in[5];
    const float* b1   = (const float*)d_in[6];
    const float* Wfc  = (const float*)d_in[7];
    const float* bfc  = (const float*)d_in[8];
    float* out = (float*)d_out;

    const int NN = in_sizes[0] / D_IN;       // 50000
    const int E  = in_sizes[1] / 2;          // 800000
    const int G  = out_size / NCLASS;        // 500

    const int* erow = eidx;                  // edge_index[0]
    const int* ecol = eidx + E;              // edge_index[1]

    // ---- workspace layout (256B aligned sections) ----
    char* ws = (char*)d_ws;
    size_t off = 0;
    auto alloc = [&](size_t bytes) -> char* {
        char* p = ws + off;
        off = (off + bytes + 255) & ~(size_t)255;
        return p;
    };
    float*    dinv = (float*)   alloc((size_t)NN * sizeof(float));
    _Float16* xh   = (_Float16*)alloc((size_t)NN * D_IN * sizeof(_Float16));
    _Float16* w0t  = (_Float16*)alloc((size_t)HIDDEN * D_IN * sizeof(_Float16));
    _Float16* w1t  = (_Float16*)alloc((size_t)HIDDEN * HIDDEN * sizeof(_Float16));
    _Float16* lin  = (_Float16*)alloc((size_t)NN * HIDDEN * sizeof(_Float16));
    float*    agg  = (float*)   alloc((size_t)NN * HIDDEN * sizeof(float));
    _Float16* hbuf = (_Float16*)alloc((size_t)NN * HIDDEN * sizeof(_Float16));
    float*    pool = (float*)   alloc((size_t)G * HIDDEN * sizeof(float));
    (void)ws_size;

    const int T = 256;
    // degree / symmetric norm
    k_deg_init <<<(NN + T - 1) / T, T, 0, stream>>>(dinv, NN);
    k_deg_edges<<<(E  + T - 1) / T, T, 0, stream>>>(erow, dinv, E);
    k_dinv     <<<(NN + T - 1) / T, T, 0, stream>>>(dinv, NN);

    // precision conversion (+ weight transpose)
    k_cvt_f16 <<<((size_t)NN * D_IN + T - 1) / T, T, 0, stream>>>(x, xh, NN * D_IN);
    k_cvt_w_t <<<(D_IN * HIDDEN + T - 1) / T, T, 0, stream>>>(W0, w0t, D_IN, HIDDEN);
    k_cvt_w_t <<<(HIDDEN * HIDDEN + T - 1) / T, T, 0, stream>>>(W1, w1t, HIDDEN, HIDDEN);

    const int mtiles  = (NN + 15) / 16;
    const int mgroups = (mtiles + 7) / 8;            // 8 M-tiles per block
    const int ngroups = HIDDEN / 64;
    const int gemmBlocks = mgroups * ngroups;

    // ---- layer 0 ----
    k_gemm_f16<<<gemmBlocks, 256, 0, stream>>>(xh, w0t, lin, NN, HIDDEN, D_IN);
    (void)hipMemsetAsync(agg, 0, (size_t)NN * HIDDEN * sizeof(float), stream);
    k_agg_edges<<<E, 128, 0, stream>>>(lin, dinv, erow, ecol, agg, E);
    k_finalize<<<NN, 256, 0, stream>>>(agg, lin, dinv, b0, hbuf, NN);

    // ---- layer 1 ----
    k_gemm_f16<<<gemmBlocks, 256, 0, stream>>>(hbuf, w1t, lin, NN, HIDDEN, HIDDEN);
    (void)hipMemsetAsync(agg, 0, (size_t)NN * HIDDEN * sizeof(float), stream);
    k_agg_edges<<<E, 128, 0, stream>>>(lin, dinv, erow, ecol, agg, E);
    k_finalize<<<NN, 256, 0, stream>>>(agg, lin, dinv, b1, hbuf, NN);

    // ---- pooling + FC ----
    (void)hipMemsetAsync(pool, 0, (size_t)G * HIDDEN * sizeof(float), stream);
    k_pool<<<NN, 256, 0, stream>>>(hbuf, ngi, pool, NN);
    k_fc<<<(G * NCLASS + T - 1) / T, T, 0, stream>>>(pool, Wfc, bfc, out, G);
}
